// GNN_5145370820834
// MI455X (gfx1250) — matrix operand order repaired and checked by reading, not compiled, not run on hardware
//
#include <hip/hip_runtime.h>

// ---------------------------------------------------------------------------
// Sizes fixed by the reference
// ---------------------------------------------------------------------------
#define N_U   50000
#define N_I   50000
#define NE    800000
#define E_LBL 200000
#define D_IN  128
#define HID   256
#define OUT_D 128

typedef __attribute__((ext_vector_type(16))) __bf16 v16bf;
typedef __attribute__((ext_vector_type(8)))  __bf16 v8bf;
typedef __attribute__((ext_vector_type(8)))  float  v8f;

// ---------------------------------------------------------------------------
// Utility kernels
// ---------------------------------------------------------------------------
__global__ void __launch_bounds__(256)
fill_zero(float* __restrict__ p, int n) {
    int i = blockIdx.x * blockDim.x + threadIdx.x;
    if (i < n) p[i] = 0.0f;
}

// f32 -> bf16, 8 elements per thread (vectorized b128 in / b128 out)
__global__ void __launch_bounds__(256)
cvt_bf16(const float* __restrict__ src, __bf16* __restrict__ dst, int n8) {
    int i = blockIdx.x * blockDim.x + threadIdx.x;
    if (i >= n8) return;
    const float4* s = reinterpret_cast<const float4*>(src);
    float4 a = s[2 * i], b = s[2 * i + 1];
    v8bf o;
    o[0] = (__bf16)a.x; o[1] = (__bf16)a.y; o[2] = (__bf16)a.z; o[3] = (__bf16)a.w;
    o[4] = (__bf16)b.x; o[5] = (__bf16)b.y; o[6] = (__bf16)b.z; o[7] = (__bf16)b.w;
    *reinterpret_cast<v8bf*>(dst + i * 8) = o;
}

__global__ void __launch_bounds__(256)
count_edges(const int* __restrict__ edge, float* __restrict__ cnt, int E) {
    int e = blockIdx.x * blockDim.x + threadIdx.x;
    if (e < E) unsafeAtomicAdd(&cnt[edge[E + e]], 1.0f);   // dst row is edge[1]
}

// agg[dst] += x[src], 4 floats per thread
__global__ void __launch_bounds__(256)
scatter_add(const float* __restrict__ x, const int* __restrict__ edge,
            float* __restrict__ agg, int E, int Dq /* = D/4 */) {
    int idx = blockIdx.x * blockDim.x + threadIdx.x;
    if (idx >= E * Dq) return;
    int e = idx / Dq;
    int c = idx - e * Dq;
    int src = edge[e];
    int dst = edge[E + e];
    const float4 v = reinterpret_cast<const float4*>(x)[src * Dq + c];
    float* p = agg + (dst * Dq + c) * 4;
    unsafeAtomicAdd(p + 0, v.x);
    unsafeAtomicAdd(p + 1, v.y);
    unsafeAtomicAdd(p + 2, v.z);
    unsafeAtomicAdd(p + 3, v.w);
}

// mean[row,:] = bf16( agg[row,:] / max(cnt[row],1) );  D = 1<<shift, 8 elems/thread
__global__ void __launch_bounds__(256)
row_mean_bf16(const float* __restrict__ agg, const float* __restrict__ cnt,
              __bf16* __restrict__ mean, int shift, int n8) {
    int i = blockIdx.x * blockDim.x + threadIdx.x;
    if (i >= n8) return;
    int base = i * 8;
    int row  = base >> shift;
    float r  = 1.0f / fmaxf(cnt[row], 1.0f);
    const float4* s = reinterpret_cast<const float4*>(agg);
    float4 a = s[2 * i], b = s[2 * i + 1];
    v8bf o;
    o[0] = (__bf16)(a.x * r); o[1] = (__bf16)(a.y * r);
    o[2] = (__bf16)(a.z * r); o[3] = (__bf16)(a.w * r);
    o[4] = (__bf16)(b.x * r); o[5] = (__bf16)(b.y * r);
    o[6] = (__bf16)(b.z * r); o[7] = (__bf16)(b.w * r);
    *reinterpret_cast<v8bf*>(mean + base) = o;
}

// ---------------------------------------------------------------------------
// Pack a row-major f32 weight [K,N] into per-lane WMMA B-fragments (bf16).
// Tile (tk,tn) = 32(K) x 16(N).  Element j of lane l holds W[tk*32+(l>>4)*8+j+(j>=8?8:0),
// tn*16+(l&15)] so the GEMM reads one contiguous 32B block per lane.
// One thread per (tile, lane).
// ---------------------------------------------------------------------------
__global__ void __launch_bounds__(256)
pack_w(const float* __restrict__ W, __bf16* __restrict__ P, int K, int N) {
    int t = blockIdx.x * blockDim.x + threadIdx.x;
    int total = (K / 32) * (N / 16) * 32;
    if (t >= total) return;
    int lane = t & 31;
    int tile = t >> 5;
    int tn = tile % (N / 16);
    int tk = tile / (N / 16);
    int col   = tn * 16 + (lane & 15);
    int kbase = tk * 32 + (lane >> 4) * 8;
    v16bf o;
#pragma unroll
    for (int j = 0; j < 16; ++j) o[j] = (__bf16)W[(kbase + j + (j >= 8 ? 8 : 0)) * N + col];
    *reinterpret_cast<v16bf*>(P + t * 16) = o;
}

// ---------------------------------------------------------------------------
// WMMA dual-GEMM:  C = A1@B1 + A2@B2 + bias   (bf16 operands, f32 accumulate)
// One wave per 16x64 tile.  Per K-step: A = two contiguous b128 loads/lane,
// then ALL four packed-B fragments are loaded into registers (constant
// +1024B offsets from a running pointer, one clause / one wait) before the
// four back-to-back WMMAs issue, so the next iteration's loads overlap the
// matrix pipe.
// ---------------------------------------------------------------------------
__device__ __forceinline__ void wmma_accum(const __bf16* __restrict__ A,
                                           const __bf16* __restrict__ Bp,
                                           int row, int tileN, int lane,
                                           int K, int Ntiles, v8f acc[4]) {
    const int half = lane >> 4;
    const __bf16* aptr = A + (size_t)row * K + half * 8;
    const __bf16* bptr = Bp + ((size_t)tileN * 32 + lane) * 16;
    const size_t ktStep = (size_t)Ntiles * 32 * 16;   // packed elems per K-step
    for (int kt = 0; kt < K / 32; ++kt) {
        v8bf lo = *reinterpret_cast<const v8bf*>(aptr);
        v8bf hi = *reinterpret_cast<const v8bf*>(aptr + 16);
        v16bf bfrag[4];
#pragma unroll
        for (int s = 0; s < 4; ++s) {
            bfrag[s] = *reinterpret_cast<const v16bf*>(bptr + s * (32 * 16));
        }
        v16bf a;
#pragma unroll
        for (int i = 0; i < 8; ++i) { a[i] = lo[i]; a[8 + i] = hi[i]; }
#pragma unroll
        for (int s = 0; s < 4; ++s) {
            acc[s] = __builtin_amdgcn_wmma_f32_16x16x32_bf16(
                false, a, false, bfrag[s], (short)0, acc[s], false, false);
        }
        aptr += 32;
        bptr += ktStep;
    }
}

__global__ void __launch_bounds__(32)
dual_gemm_wmma(float* __restrict__ Cf, __bf16* __restrict__ Cbf,
               const __bf16* __restrict__ A1, const __bf16* __restrict__ B1p,
               const __bf16* __restrict__ A2, const __bf16* __restrict__ B2p,
               const float* __restrict__ bias, int K, int N) {
    const int lane  = threadIdx.x & 31;
    const int row   = blockIdx.x * 16 + (lane & 15);
    const int tileN = blockIdx.y * 4;

    v8f acc[4] = {};
    wmma_accum(A1, B1p, row, tileN, lane, K, N / 16, acc);
    wmma_accum(A2, B2p, row, tileN, lane, K, N / 16, acc);

    const int half    = lane >> 4;
    const int lan     = lane & 15;
    const int rowBase = blockIdx.x * 16 + half * 8;
#pragma unroll
    for (int s = 0; s < 4; ++s) {
        const int col  = (tileN + s) * 16 + lan;
        const float bv = bias[col];
#pragma unroll
        for (int r = 0; r < 8; ++r) {
            float v = acc[s][r] + bv;
            if (Cf)  Cf[(rowBase + r) * N + col]  = v;
            if (Cbf) Cbf[(rowBase + r) * N + col] = (__bf16)v;
        }
    }
}

// ---------------------------------------------------------------------------
// Decoder: pred[e] = dot(z_user[edge[0,e]], z_item[edge[1,e]]) over 128 dims.
// One wave32 per edge, one float4 per lane, shuffle reduction.
// ---------------------------------------------------------------------------
__global__ void __launch_bounds__(256)
decode_dot(const float* __restrict__ zu, const float* __restrict__ zi,
           const int* __restrict__ edge, float* __restrict__ out, int E) {
    int gid  = blockIdx.x * blockDim.x + threadIdx.x;
    int wave = gid >> 5;
    int lane = gid & 31;
    if (wave >= E) return;
    int u  = edge[wave];
    int it = edge[E + wave];
    const float4 a = reinterpret_cast<const float4*>(zu)[u * 32 + lane];
    const float4 b = reinterpret_cast<const float4*>(zi)[it * 32 + lane];
    float s = a.x * b.x + a.y * b.y + a.z * b.z + a.w * b.w;
#pragma unroll
    for (int off = 16; off > 0; off >>= 1) s += __shfl_xor(s, off, 32);
    if (lane == 0) out[wave] = s;
}

// ---------------------------------------------------------------------------
// Host orchestration (all on `stream`, graph-capture safe)
// ---------------------------------------------------------------------------
extern "C" void kernel_launch(void* const* d_in, const int* in_sizes, int n_in,
                              void* d_out, int out_size, void* d_ws, size_t ws_size,
                              hipStream_t stream) {
    const float* x_user     = (const float*)d_in[0];
    const float* x_item     = (const float*)d_in[1];
    const int*   edge_ui    = (const int*)d_in[2];
    const int*   edge_iu    = (const int*)d_in[3];
    const int*   edge_lbl   = (const int*)d_in[4];
    const float* w_msg_ui0  = (const float*)d_in[5];
    const float* b_ui0      = (const float*)d_in[6];
    const float* w_root_ui0 = (const float*)d_in[7];
    const float* w_msg_iu0  = (const float*)d_in[8];
    const float* b_iu0      = (const float*)d_in[9];
    const float* w_root_iu0 = (const float*)d_in[10];
    const float* w_msg_ui1  = (const float*)d_in[11];
    const float* b_ui1      = (const float*)d_in[12];
    const float* w_root_ui1 = (const float*)d_in[13];
    const float* w_msg_iu1  = (const float*)d_in[14];
    const float* b_iu1      = (const float*)d_in[15];
    const float* w_root_iu1 = (const float*)d_in[16];
    const float* w_lin_u    = (const float*)d_in[17];
    const float* b_lin_u    = (const float*)d_in[18];
    const float* w_lin_i    = (const float*)d_in[19];
    const float* b_lin_i    = (const float*)d_in[20];
    float* pred = (float*)d_out;

    // ---- workspace carve-up (bytes, ~411 MB with overlays) ----
    char* p = (char*)d_ws;
    auto carve = [&](size_t bytes) { char* r = p; p += (bytes + 255) & ~size_t(255); return r; };

    float*  cnt_i   = (float*)carve(N_I * 4);
    float*  cnt_u   = (float*)carve(N_U * 4);
    float*  agg0_i  = (float*)carve((size_t)N_I * D_IN * 4);
    float*  agg0_u  = (float*)carve((size_t)N_U * D_IN * 4);
    float*  agg1_i  = (float*)carve((size_t)N_I * HID * 4);
    float*  agg1_u  = (float*)carve((size_t)N_U * HID * 4);
    float*  item0_f = (float*)carve((size_t)N_I * HID * 4);
    float*  user0_f = (float*)carve((size_t)N_U * HID * 4);
    __bf16* xu_bf   = (__bf16*)carve((size_t)N_U * D_IN * 2);
    __bf16* xi_bf   = (__bf16*)carve((size_t)N_I * D_IN * 2);
    __bf16* mean0_i = (__bf16*)carve((size_t)N_I * D_IN * 2);
    __bf16* mean0_u = (__bf16*)carve((size_t)N_U * D_IN * 2);
    __bf16* mean1_i = (__bf16*)carve((size_t)N_I * HID * 2);
    __bf16* mean1_u = (__bf16*)carve((size_t)N_U * HID * 2);
    __bf16* item0_bf = (__bf16*)carve((size_t)N_I * HID * 2);
    __bf16* user0_bf = (__bf16*)carve((size_t)N_U * HID * 2);
    // packed weights (bf16, K*N each)
    __bf16* p_msg_ui0  = (__bf16*)carve(D_IN * HID * 2);
    __bf16* p_root_ui0 = (__bf16*)carve(D_IN * HID * 2);
    __bf16* p_msg_iu0  = (__bf16*)carve(D_IN * HID * 2);
    __bf16* p_root_iu0 = (__bf16*)carve(D_IN * HID * 2);
    __bf16* p_msg_ui1  = (__bf16*)carve(HID * HID * 2);
    __bf16* p_root_ui1 = (__bf16*)carve(HID * HID * 2);
    __bf16* p_msg_iu1  = (__bf16*)carve(HID * HID * 2);
    __bf16* p_root_iu1 = (__bf16*)carve(HID * HID * 2);
    __bf16* p_lin_u0   = (__bf16*)carve(HID * OUT_D * 2);
    __bf16* p_lin_u1   = (__bf16*)carve(HID * OUT_D * 2);
    __bf16* p_lin_i0   = (__bf16*)carve(HID * OUT_D * 2);
    __bf16* p_lin_i1   = (__bf16*)carve(HID * OUT_D * 2);
    // overlays into freed regions:
    __bf16* item1_bf = (__bf16*)agg1_i;                       // after row_mean layer1
    __bf16* user1_bf = (__bf16*)((char*)agg1_i + (size_t)N_I * HID * 2);
    float*  z_user   = agg0_i;                                // after row_mean layer0
    float*  z_item   = agg0_u;

    const int TB = 256;
    auto blk = [&](long n) { return (int)((n + TB - 1) / TB); };

    // ---- one-time prep: pack weights, convert inputs ----
    pack_w<<<blk((D_IN/32)*(HID/16)*32), TB, 0, stream>>>(w_msg_ui0,  p_msg_ui0,  D_IN, HID);
    pack_w<<<blk((D_IN/32)*(HID/16)*32), TB, 0, stream>>>(w_root_ui0, p_root_ui0, D_IN, HID);
    pack_w<<<blk((D_IN/32)*(HID/16)*32), TB, 0, stream>>>(w_msg_iu0,  p_msg_iu0,  D_IN, HID);
    pack_w<<<blk((D_IN/32)*(HID/16)*32), TB, 0, stream>>>(w_root_iu0, p_root_iu0, D_IN, HID);
    pack_w<<<blk((HID/32)*(HID/16)*32),  TB, 0, stream>>>(w_msg_ui1,  p_msg_ui1,  HID, HID);
    pack_w<<<blk((HID/32)*(HID/16)*32),  TB, 0, stream>>>(w_root_ui1, p_root_ui1, HID, HID);
    pack_w<<<blk((HID/32)*(HID/16)*32),  TB, 0, stream>>>(w_msg_iu1,  p_msg_iu1,  HID, HID);
    pack_w<<<blk((HID/32)*(HID/16)*32),  TB, 0, stream>>>(w_root_iu1, p_root_iu1, HID, HID);
    pack_w<<<blk((HID/32)*(OUT_D/16)*32), TB, 0, stream>>>(w_lin_u,                p_lin_u0, HID, OUT_D);
    pack_w<<<blk((HID/32)*(OUT_D/16)*32), TB, 0, stream>>>(w_lin_u + HID * OUT_D,  p_lin_u1, HID, OUT_D);
    pack_w<<<blk((HID/32)*(OUT_D/16)*32), TB, 0, stream>>>(w_lin_i,                p_lin_i0, HID, OUT_D);
    pack_w<<<blk((HID/32)*(OUT_D/16)*32), TB, 0, stream>>>(w_lin_i + HID * OUT_D,  p_lin_i1, HID, OUT_D);
    cvt_bf16<<<blk(N_U * D_IN / 8), TB, 0, stream>>>(x_user, xu_bf, N_U * D_IN / 8);
    cvt_bf16<<<blk(N_I * D_IN / 8), TB, 0, stream>>>(x_item, xi_bf, N_I * D_IN / 8);

    // ---- layer 0 ----
    const int n_zero0 = N_I + N_U + N_I * D_IN + N_U * D_IN;   // cnt+agg0 contiguous
    fill_zero<<<blk(n_zero0), TB, 0, stream>>>(cnt_i, n_zero0);
    count_edges<<<blk(NE), TB, 0, stream>>>(edge_ui, cnt_i, NE);
    count_edges<<<blk(NE), TB, 0, stream>>>(edge_iu, cnt_u, NE);
    scatter_add<<<blk((long)NE * (D_IN/4)), TB, 0, stream>>>(x_user, edge_ui, agg0_i, NE, D_IN/4);
    scatter_add<<<blk((long)NE * (D_IN/4)), TB, 0, stream>>>(x_item, edge_iu, agg0_u, NE, D_IN/4);
    row_mean_bf16<<<blk(N_I * D_IN / 8), TB, 0, stream>>>(agg0_i, cnt_i, mean0_i, 7, N_I * D_IN / 8);
    row_mean_bf16<<<blk(N_U * D_IN / 8), TB, 0, stream>>>(agg0_u, cnt_u, mean0_u, 7, N_U * D_IN / 8);
    dual_gemm_wmma<<<dim3(N_I/16, HID/64), 32, 0, stream>>>(item0_f, item0_bf, mean0_i, p_msg_ui0, xi_bf, p_root_ui0, b_ui0, D_IN, HID);
    dual_gemm_wmma<<<dim3(N_U/16, HID/64), 32, 0, stream>>>(user0_f, user0_bf, mean0_u, p_msg_iu0, xu_bf, p_root_iu0, b_iu0, D_IN, HID);

    // ---- layer 1 ----
    const int n_zero1 = N_I * HID + N_U * HID;                 // agg1 contiguous
    fill_zero<<<blk(n_zero1), TB, 0, stream>>>(agg1_i, n_zero1);
    scatter_add<<<blk((long)NE * (HID/4)), TB, 0, stream>>>(user0_f, edge_ui, agg1_i, NE, HID/4);
    scatter_add<<<blk((long)NE * (HID/4)), TB, 0, stream>>>(item0_f, edge_iu, agg1_u, NE, HID/4);
    row_mean_bf16<<<blk(N_I * HID / 8), TB, 0, stream>>>(agg1_i, cnt_i, mean1_i, 8, N_I * HID / 8);
    row_mean_bf16<<<blk(N_U * HID / 8), TB, 0, stream>>>(agg1_u, cnt_u, mean1_u, 8, N_U * HID / 8);
    dual_gemm_wmma<<<dim3(N_I/16, HID/64), 32, 0, stream>>>((float*)nullptr, item1_bf, mean1_i, p_msg_ui1, item0_bf, p_root_ui1, b_ui1, HID, HID);
    dual_gemm_wmma<<<dim3(N_U/16, HID/64), 32, 0, stream>>>((float*)nullptr, user1_bf, mean1_u, p_msg_iu1, user0_bf, p_root_iu1, b_iu1, HID, HID);

    // ---- JK concat + per-type linear:  z = h0 @ W[0:256,:] + h1 @ W[256:512,:] + b ----
    dual_gemm_wmma<<<dim3(N_U/16, OUT_D/64), 32, 0, stream>>>(z_user, (__bf16*)nullptr, user0_bf, p_lin_u0, user1_bf, p_lin_u1, b_lin_u, HID, OUT_D);
    dual_gemm_wmma<<<dim3(N_I/16, OUT_D/64), 32, 0, stream>>>(z_item, (__bf16*)nullptr, item0_bf, p_lin_i0, item1_bf, p_lin_i1, b_lin_i, HID, OUT_D);

    // ---- decode ----
    decode_dot<<<blk((long)E_LBL * 32), TB, 0, stream>>>(z_user, z_item, edge_lbl, pred, E_LBL);
}
